// GroupedQueryAttention_71545565217173
// MI455X (gfx1250) — compile-verified, matrix operand order checked by reading, not run on hardware
//
#include <hip/hip_runtime.h>
#include <hip/hip_bf16.h>

// ---------------------------------------------------------------------------
// MI455X (gfx1250) GQA implementation.
// Precision strategy: bf16 operands + f32 WMMA accumulation everywhere
// (v_wmma_f32_16x16x32_bf16 gives 8x the K-depth per issue of the f32 WMMA).
// Data movement: Tensor Data Mover stages the non-transposed LDS tiles
// (tracked with TENSORcnt); transposed tiles are staged manually.
// ---------------------------------------------------------------------------

typedef __attribute__((ext_vector_type(16))) __bf16 v16bf;
typedef __attribute__((ext_vector_type(8)))  float  v8f;

#define WMMA_BF16(a, b, c) \
    __builtin_amdgcn_wmma_f32_16x16x32_bf16(false, (a), false, (b), (short)0, (c), false, false)

#if __has_builtin(__builtin_amdgcn_tensor_load_to_lds)
#define HAVE_TDM 1
#else
#define HAVE_TDM 0
#endif

#if HAVE_TDM
// ---------------------------------------------------------------------------
// Issue one TDM 2D tile load (global -> LDS), bf16 elements.
//   lds_off          : LDS byte offset of destination
//   gptr             : global address of tile start
//   w_elems          : tile width in elements (contiguous dim)
//   h_rows           : tile height in rows
//   row_stride_elems : global stride between rows, in elements
// D# layout per CDNA5 ISA ch.8 (group0 128b, group1 256b, groups 2/3 zero).
// ---------------------------------------------------------------------------
__device__ __forceinline__ void tdm_load_2d_bf16(unsigned lds_off, const void* gptr,
                                                 unsigned w_elems, unsigned h_rows,
                                                 unsigned row_stride_elems) {
  typedef unsigned u32x4 __attribute__((ext_vector_type(4)));
  typedef int      i32x4 __attribute__((ext_vector_type(4)));
  typedef int      i32x8 __attribute__((ext_vector_type(8)));

  const unsigned long long ga = (unsigned long long)(size_t)gptr;

  union { unsigned u[4]; u32x4 v; } g0;
  g0.u[0] = 1u;                                            // count=1 (valid, user mode)
  g0.u[1] = lds_off;                                       // lds_addr
  g0.u[2] = (unsigned)ga;                                  // global_addr[31:0]
  g0.u[3] = (unsigned)((ga >> 32) & 0x1FFFFFFu)            // global_addr[56:32]
          | (2u << 30);                                    // type = 2 ("image")

  union { unsigned long long q[4]; i32x8 v; } g1;
  g1.q[0] = (1ull << 16)                                   // data_size = 2 bytes
          | ((unsigned long long)(w_elems & 0xFFFFu) << 48);        // tensor_dim0[15:0]
  g1.q[1] = (unsigned long long)(w_elems >> 16)                     // tensor_dim0[31:16]
          | ((unsigned long long)h_rows << 16)                      // tensor_dim1 (111:80)
          | ((unsigned long long)(w_elems & 0xFFFFu) << 48);        // tile_dim0 (127:112)
  g1.q[2] = (unsigned long long)(h_rows & 0xFFFFu)                  // tile_dim1 (143:128)
          | ((unsigned long long)row_stride_elems << 32);           // tensor_dim0_stride[31:0] (191:160)
  g1.q[3] = 0ull;                                                   // stride[47:32]=0, dim1_stride=0

  i32x4 z4 = {0, 0, 0, 0};
#if defined(__clang_major__) && (__clang_major__ >= 23)
  i32x8 z8 = {0, 0, 0, 0, 0, 0, 0, 0};
  __builtin_amdgcn_tensor_load_to_lds(g0.v, g1.v, z4, z4, z8, 0);
#else
  __builtin_amdgcn_tensor_load_to_lds(g0.v, g1.v, z4, z4, 0);
#endif
}
#endif  // HAVE_TDM

__device__ __forceinline__ void wait_tensorcnt0() {
#if __has_builtin(__builtin_amdgcn_s_wait_tensorcnt)
  __builtin_amdgcn_s_wait_tensorcnt(0);
#endif
}

// ---------------------------------------------------------------------------
// fp32 -> bf16 conversion
// ---------------------------------------------------------------------------
__global__ __launch_bounds__(256) void cvt_f32_bf16(const float* __restrict__ in,
                                                    __bf16* __restrict__ out, int n) {
  int i = blockIdx.x * 256 + threadIdx.x;
  if (i < n) out[i] = (__bf16)in[i];
}

// ---------------------------------------------------------------------------
// Tiled WMMA GEMM: C[M,N] = A[M,K] * B[K,N], A/B bf16, C f32 or bf16.
// WG = 256 threads = 8 waves; WG tile 128x128; wave tile 32x64 (2x4 accums).
// A tile (128x32, row-major) staged by TDM; B tile staged manually transposed
// [n][k] so both fragment loads are contiguous 32B per lane.
// ---------------------------------------------------------------------------
template <bool OUT_BF16>
__global__ __launch_bounds__(256) void gemm_wmma(const __bf16* __restrict__ A,
                                                 const __bf16* __restrict__ B,
                                                 void* __restrict__ Cv,
                                                 int M, int N, int K) {
  __shared__ __bf16 sA[128 * 32];
  __shared__ __bf16 sB[128 * 32];   // transposed: [n][k]

  const int tid  = threadIdx.x;
  const int lane = tid & 31;
  const int wid  = tid >> 5;        // 0..7 (wave32)
  const int half = lane >> 4;       // 0/1 : K-halves of WMMA fragments
  const int l16  = lane & 15;

  const int tileM = blockIdx.y * 128;
  const int tileN = blockIdx.x * 128;
  const int waveM = (wid >> 1) * 32;   // 0,32,64,96
  const int waveN = (wid & 1) * 64;    // 0,64

  v8f acc[2][4];
  v8f vzero = {};
#pragma unroll
  for (int mt = 0; mt < 2; ++mt)
#pragma unroll
    for (int nt = 0; nt < 4; ++nt) acc[mt][nt] = vzero;

  // staging indices
#if !HAVE_TDM
  const int arow = tid >> 1;         // 0..127
  const int akk  = (tid & 1) * 16;   // 0/16
#endif
  const int bk   = tid >> 3;         // 0..31
  const int bnn  = (tid & 7) * 16;   // 0..112

  for (int k0 = 0; k0 < K; k0 += 32) {
    __syncthreads();   // previous iteration's LDS readers done
    // ---- stage A (row-major 128x32) ----
#if HAVE_TDM
    if (tid == 0)
      tdm_load_2d_bf16((unsigned)(size_t)(void*)sA,
                       A + (size_t)tileM * K + k0,
                       /*w*/32, /*h*/128, /*row stride*/(unsigned)K);
#else
    {
      const __bf16* ga = A + (size_t)(tileM + arow) * K + k0 + akk;
      uint4 a0 = ((const uint4*)ga)[0];
      uint4 a1 = ((const uint4*)ga)[1];
      ((uint4*)&sA[arow * 32 + akk])[0] = a0;
      ((uint4*)&sA[arow * 32 + akk])[1] = a1;
    }
#endif
    // ---- stage B transposed ----
    const __bf16* gb = B + (size_t)(k0 + bk) * N + tileN + bnn;
    __bf16 tb[16];
    *(uint4*)&tb[0] = ((const uint4*)gb)[0];
    *(uint4*)&tb[8] = ((const uint4*)gb)[1];
#pragma unroll
    for (int i = 0; i < 16; ++i) sB[(bnn + i) * 32 + bk] = tb[i];
    // prefetch next K tile (global_prefetch_b8)
    if (k0 + 32 < K)
      __builtin_prefetch(B + (size_t)(k0 + 32 + bk) * N + tileN + bnn, 0, 0);
    wait_tensorcnt0();
    __syncthreads();

    // ---- fragments: lanes 0-15 hold K=0-15, lanes 16-31 hold K=16-31 ----
    v16bf af[2], bfr[4];
#pragma unroll
    for (int mt = 0; mt < 2; ++mt)
      af[mt] = *(const v16bf*)&sA[(waveM + mt * 16 + l16) * 32 + half * 16];
#pragma unroll
    for (int nt = 0; nt < 4; ++nt)
      bfr[nt] = *(const v16bf*)&sB[(waveN + nt * 16 + l16) * 32 + half * 16];

#pragma unroll
    for (int mt = 0; mt < 2; ++mt)
#pragma unroll
      for (int nt = 0; nt < 4; ++nt)
        acc[mt][nt] = WMMA_BF16(af[mt], bfr[nt], acc[mt][nt]);
  }

  // ---- store: C layout row = r + 8*half (per vgpr), col = l16 ----
#pragma unroll
  for (int mt = 0; mt < 2; ++mt) {
#pragma unroll
    for (int nt = 0; nt < 4; ++nt) {
#pragma unroll
      for (int r = 0; r < 8; ++r) {
        const int row = tileM + waveM + mt * 16 + half * 8 + r;
        const int col = tileN + waveN + nt * 16 + l16;
        const float val = acc[mt][nt][r];
        if (OUT_BF16)
          ((__bf16*)Cv)[(size_t)row * N + col] = (__bf16)val;
        else
          ((float*)Cv)[(size_t)row * N + col] = val;
      }
    }
  }
}

// ---------------------------------------------------------------------------
// RoPE in place on bf16 tensor laid out [B*T, H, 128]. One thread per (bt,h,i)
// rotates the pair (i, i+64).  inv_freq = 10000^(-i/64) = exp(-i*ln(1e4)/64).
// ---------------------------------------------------------------------------
__global__ __launch_bounds__(256) void rope_kernel(__bf16* __restrict__ t, int H, int total) {
  int idx = blockIdx.x * 256 + threadIdx.x;
  if (idx >= total) return;
  const int i   = idx & 63;
  int tmp       = idx >> 6;
  const int h   = tmp % H;
  const int bt  = tmp / H;
  const int pos = bt & 2047;   // T = 2048
  const float ang = (float)pos * __expf(-0.14391156831212787f * (float)i);
  float s, c;
  __sincosf(ang, &s, &c);
  __bf16* p = t + ((size_t)bt * H + h) * 128;
  const float x1 = (float)p[i];
  const float x2 = (float)p[i + 64];
  p[i]      = (__bf16)(x1 * c - x2 * s);
  p[i + 64] = (__bf16)(x2 * c + x1 * s);
}

// ---------------------------------------------------------------------------
// Flash attention, sliding window 512, GQA (16 q heads, 4 kv heads).
// WG = 128 threads = 4 waves; each wave owns 16 queries; WG covers 64 queries
// of one (b,h). Per iteration the WG stages 32 keys x 128 dims of K (TDM,
// row major) and V (manual, transposed) into LDS, then each wave:
//   S = Q K^T           : 8x v_wmma_f32_16x16x32_bf16
//   online softmax      : shfl_xor reductions inside 16-lane halves
//   P restage via LDS   : C-layout -> A-layout transpose
//   O += P V            : 8x v_wmma_f32_16x16x32_bf16
// ---------------------------------------------------------------------------
__global__ __launch_bounds__(128) void attn_kernel(const __bf16* __restrict__ Q,
                                                   const __bf16* __restrict__ Kc,
                                                   const __bf16* __restrict__ Vc,
                                                   __bf16* __restrict__ O) {
  constexpr int T = 2048, HQ = 16, HKV = 4, HD = 128, WIN = 512;
  __shared__ __bf16 sK[32 * 128];     // [key][d]
  __shared__ __bf16 sVt[128 * 32];    // [d][key]
  __shared__ __bf16 sP[4][16 * 32];   // per-wave P staging [row][key]

  const int tid  = threadIdx.x;
  const int lane = tid & 31;
  const int wid  = tid >> 5;          // 0..3
  const int half = lane >> 4;
  const int l16  = lane & 15;

  const int b    = blockIdx.y >> 4;
  const int h    = blockIdx.y & 15;
  const int kvh  = h >> 2;
  const int q0wg = blockIdx.x * 64;
  const int q0w  = q0wg + wid * 16;

  // ---- Q fragments (A-layout: lane -> row l16, 16 contiguous d per half) ----
  v16bf qf[4];
  {
    const int qrow = q0w + l16;
    const __bf16* gq = Q + ((size_t)(b * T + qrow) * HQ + h) * HD;
#pragma unroll
    for (int kk = 0; kk < 4; ++kk)
      qf[kk] = *(const v16bf*)(gq + kk * 32 + half * 16);
  }

  v8f oacc[8];
  v8f vzero = {};
#pragma unroll
  for (int i = 0; i < 8; ++i) oacc[i] = vzero;
  float mrun[8], lrun[8];
#pragma unroll
  for (int i = 0; i < 8; ++i) { mrun[i] = -1e30f; lrun[i] = 0.0f; }

  const float scale = 0.08838834764831845f;   // 1/sqrt(128)

  int kb_lo = q0wg - 512;
  if (kb_lo < 0) kb_lo = 0;                   // q0wg is 64-aligned -> 32-aligned
  const int kb_hi = q0wg + 63;

  // V staging: 128 threads cover 32x128 tile, 32 elems (64B) each
  const int skey = tid >> 2;        // 0..31
  const int sd   = (tid & 3) * 32;  // 0,32,64,96

  for (int kb = kb_lo; kb <= kb_hi; kb += 32) {
    __syncthreads();   // previous iteration's readers done
    // ---- stage K (row-major 32x128) via Tensor Data Mover ----
#if HAVE_TDM
    if (tid == 0)
      tdm_load_2d_bf16((unsigned)(size_t)(void*)sK,
                       Kc + ((size_t)(b * T + kb) * HKV + kvh) * HD,
                       /*w*/HD, /*h*/32, /*row stride*/HKV * HD);
#else
    {
      const __bf16* gk = Kc + ((size_t)(b * T + kb + skey) * HKV + kvh) * HD + sd;
      uint4 k0 = ((const uint4*)gk)[0], k1 = ((const uint4*)gk)[1],
            k2 = ((const uint4*)gk)[2], k3 = ((const uint4*)gk)[3];
      ((uint4*)&sK[skey * 128 + sd])[0] = k0;
      ((uint4*)&sK[skey * 128 + sd])[1] = k1;
      ((uint4*)&sK[skey * 128 + sd])[2] = k2;
      ((uint4*)&sK[skey * 128 + sd])[3] = k3;
    }
#endif
    // ---- stage V transposed (manual) ----
    {
      const __bf16* gv = Vc + ((size_t)(b * T + kb + skey) * HKV + kvh) * HD + sd;
      __bf16 tv[32];
      *(uint4*)&tv[0]  = ((const uint4*)gv)[0];
      *(uint4*)&tv[8]  = ((const uint4*)gv)[1];
      *(uint4*)&tv[16] = ((const uint4*)gv)[2];
      *(uint4*)&tv[24] = ((const uint4*)gv)[3];
#pragma unroll
      for (int i = 0; i < 32; ++i) sVt[(sd + i) * 32 + skey] = tv[i];
      if (kb + 32 <= kb_hi)
        __builtin_prefetch(gv + 32 * HKV * HD, 0, 0);
    }
    wait_tensorcnt0();
    __syncthreads();   // K/V tile visible

    // ---- scores S = Q K^T : two 16-key sub-tiles ----
    v8f sc[2];
#pragma unroll
    for (int sub = 0; sub < 2; ++sub) {
      sc[sub] = vzero;
#pragma unroll
      for (int kk = 0; kk < 4; ++kk) {
        v16bf kf = *(const v16bf*)&sK[(sub * 16 + l16) * 128 + kk * 32 + half * 16];
        sc[sub] = WMMA_BF16(qf[kk], kf, sc[sub]);
      }
    }

    // ---- scale + window mask (row = r + 8*half, col key = l16) ----
#pragma unroll
    for (int sub = 0; sub < 2; ++sub) {
      const int key = kb + sub * 16 + l16;
#pragma unroll
      for (int r = 0; r < 8; ++r) {
        const int qq = q0w + half * 8 + r;
        const float s = sc[sub][r] * scale;
        const bool keep = (key <= qq) && (qq - key < WIN);
        sc[sub][r] = keep ? s : -1e30f;
      }
    }

    // ---- online softmax + P staging ----
#pragma unroll
    for (int r = 0; r < 8; ++r) {
      const float v0 = sc[0][r], v1 = sc[1][r];
      float mx = fmaxf(v0, v1);
#pragma unroll
      for (int off = 8; off >= 1; off >>= 1)
        mx = fmaxf(mx, __shfl_xor(mx, off, 32));
      const float mo = mrun[r];
      const float mn = fmaxf(mo, mx);
      const float alpha = __expf(mo - mn);
      const float p0 = (v0 > -1e29f) ? __expf(v0 - mn) : 0.0f;
      const float p1 = (v1 > -1e29f) ? __expf(v1 - mn) : 0.0f;
      float rs = p0 + p1;
#pragma unroll
      for (int off = 8; off >= 1; off >>= 1)
        rs += __shfl_xor(rs, off, 32);
      lrun[r] = lrun[r] * alpha + rs;
      mrun[r] = mn;
#pragma unroll
      for (int dt = 0; dt < 8; ++dt) oacc[dt][r] *= alpha;
      const int prow = half * 8 + r;
      sP[wid][prow * 32 + l16]      = (__bf16)p0;
      sP[wid][prow * 32 + 16 + l16] = (__bf16)p1;
    }
    __syncthreads();   // P visible (also keeps waves in lockstep for sK/sVt reuse)

    // ---- O += P V ----
    v16bf pf = *(const v16bf*)&sP[wid][l16 * 32 + half * 16];
#pragma unroll
    for (int dt = 0; dt < 8; ++dt) {
      v16bf vf = *(const v16bf*)&sVt[(dt * 16 + l16) * 32 + half * 16];
      oacc[dt] = WMMA_BF16(pf, vf, oacc[dt]);
    }
  }

  // ---- normalize and store [b, t, h*128 + d] (bf16 for output GEMM) ----
#pragma unroll
  for (int dt = 0; dt < 8; ++dt) {
    const int col = dt * 16 + l16;
#pragma unroll
    for (int r = 0; r < 8; ++r) {
      const int qq = q0w + half * 8 + r;
      const float val = oacc[dt][r] / lrun[r];
      O[(size_t)(b * T + qq) * 2048 + h * HD + col] = (__bf16)val;
    }
  }
}

// ---------------------------------------------------------------------------
// Host-side launcher
// ---------------------------------------------------------------------------
extern "C" void kernel_launch(void* const* d_in, const int* in_sizes, int n_in,
                              void* d_out, int out_size, void* d_ws, size_t ws_size,
                              hipStream_t stream) {
  (void)in_sizes; (void)n_in; (void)out_size; (void)ws_size;
  constexpr int BATCH = 2, T = 2048, D = 2048, HQ = 16, HKV = 4, HD = 128;
  constexpr int M = BATCH * T;                 // 4096
  constexpr int NQ = HQ * HD;                  // 2048
  constexpr int NKV = HKV * HD;                // 512

  const float* x  = (const float*)d_in[0];
  const float* wq = (const float*)d_in[1];
  const float* wk = (const float*)d_in[2];
  const float* wv = (const float*)d_in[3];
  const float* wo = (const float*)d_in[4];
  float* out = (float*)d_out;

  // workspace layout (bf16 buffers), byte offsets
  char* ws = (char*)d_ws;
  __bf16* xb  = (__bf16*)(ws);                          // 4096*2048
  __bf16* wqb = (__bf16*)(ws + 16777216);               // 2048*2048
  __bf16* wkb = (__bf16*)(ws + 25165824);               // 2048*512
  __bf16* wvb = (__bf16*)(ws + 27262976);               // 2048*512
  __bf16* wob = (__bf16*)(ws + 29360128);               // 2048*2048
  __bf16* qb  = (__bf16*)(ws + 37748736);               // 4096*2048
  __bf16* kb  = (__bf16*)(ws + 54525952);               // 4096*512
  __bf16* vb  = (__bf16*)(ws + 58720256);               // 4096*512
  __bf16* aob = (__bf16*)(ws + 62914560);               // 4096*2048

  // 1) fp32 -> bf16 conversions
  cvt_f32_bf16<<<(M * D + 255) / 256, 256, 0, stream>>>(x,  xb,  M * D);
  cvt_f32_bf16<<<(D * NQ + 255) / 256, 256, 0, stream>>>(wq, wqb, D * NQ);
  cvt_f32_bf16<<<(D * NKV + 255) / 256, 256, 0, stream>>>(wk, wkb, D * NKV);
  cvt_f32_bf16<<<(D * NKV + 255) / 256, 256, 0, stream>>>(wv, wvb, D * NKV);
  cvt_f32_bf16<<<(NQ * D + 255) / 256, 256, 0, stream>>>(wo, wob, NQ * D);

  // 2) QKV projections (WMMA GEMMs, bf16 out)
  gemm_wmma<true><<<dim3(NQ / 128,  M / 128), 256, 0, stream>>>(xb, wqb, qb, M, NQ,  D);
  gemm_wmma<true><<<dim3(NKV / 128, M / 128), 256, 0, stream>>>(xb, wkb, kb, M, NKV, D);
  gemm_wmma<true><<<dim3(NKV / 128, M / 128), 256, 0, stream>>>(xb, wvb, vb, M, NKV, D);

  // 3) RoPE on Q and K
  {
    const int tq = M * HQ * 64;
    rope_kernel<<<(tq + 255) / 256, 256, 0, stream>>>(qb, HQ, tq);
    const int tk = M * HKV * 64;
    rope_kernel<<<(tk + 255) / 256, 256, 0, stream>>>(kb, HKV, tk);
  }

  // 4) sliding-window GQA flash attention (WMMA + TDM)
  attn_kernel<<<dim3(T / 64, BATCH * HQ), 128, 0, stream>>>(qb, kb, vb, aob);

  // 5) output projection (WMMA GEMM, f32 out)
  gemm_wmma<false><<<dim3(D / 128, M / 128), 256, 0, stream>>>(aob, wob, out, M, D, NQ);
}
